// AVPViT_88038239633918
// MI455X (gfx1250) — matrix operand order repaired and checked by reading, not compiled, not run on hardware
//
#include <hip/hip_runtime.h>
#include <hip/hip_bf16.h>
#include <math.h>
#include <stdint.h>

// ---------------------------------------------------------------------------
// Model dims
// ---------------------------------------------------------------------------
static constexpr int D   = 768;
static constexpr int NH  = 12;
static constexpr int HD  = 64;
static constexpr int LNUM = 12;
static constexpr int B_  = 64;
static constexpr int NL  = 49;    // 7*7 local tokens
static constexpr int NS  = 256;   // 16*16 scene tokens
static constexpr int ML  = B_ * NL;   // 3136
static constexpr int MH  = B_ * NS;   // 16384

// GEMM tiling
static constexpr int BM = 128;
static constexpr int BN = 64;
static constexpr int BK = 32;

typedef __attribute__((ext_vector_type(16))) __bf16 v16bf;
typedef __attribute__((ext_vector_type(8)))  __bf16 v8bf;
typedef __attribute__((ext_vector_type(4)))  __bf16 v4bf;
typedef __attribute__((ext_vector_type(8)))  float  v8f;
typedef __attribute__((ext_vector_type(4)))  float  v4f;
typedef int v4i __attribute__((vector_size(16)));   // matches builtin param type

#define AS1 __attribute__((address_space(1)))
#define AS3 __attribute__((address_space(3)))

#if __has_builtin(__builtin_amdgcn_global_load_async_to_lds_b128)
#define HAVE_ASYNC_LDS 1
#else
#define HAVE_ASYNC_LDS 0
#endif

__device__ __forceinline__ void async_wait_all() {
#if HAVE_ASYNC_LDS
#if __has_builtin(__builtin_amdgcn_s_wait_asynccnt)
    __builtin_amdgcn_s_wait_asynccnt(0);
#else
    asm volatile("s_wait_asynccnt 0x0" ::: "memory");
#endif
#endif
}

// Copy 16 bytes global -> LDS (async on gfx1250 when available).
__device__ __forceinline__ void copy16_g2l(float* lds_dst, const float* gsrc) {
#if HAVE_ASYNC_LDS
    __builtin_amdgcn_global_load_async_to_lds_b128(
        (AS1 v4i*)gsrc, (AS3 v4i*)lds_dst, 0, 0);
#else
    *(v4f*)lds_dst = *(const v4f*)gsrc;
#endif
}

// ---------------------------------------------------------------------------
// GEMM: C[M,N] = epilogue( A[M,K] @ W[K,N] )
//   bias  != nullptr : acc += bias[col]
//   flags & 1        : acc = gelu_tanh(acc)
//   resid != nullptr : acc = resid[row,col] + (gate ? gate[col] : 1) * acc
// bf16 staged in LDS via vector loads; v_wmma_f32_16x16x32_bf16, f32 acc.
// Register-prefetch pipeline: global loads of tile k+1 overlap WMMA on tile k.
// ---------------------------------------------------------------------------
__global__ __launch_bounds__(256) void k_gemm_bf16(
    const float* __restrict__ A, const float* __restrict__ W, float* __restrict__ C,
    const float* __restrict__ bias, const float* __restrict__ gate,
    const float* __restrict__ resid, int flags, int M, int N, int K)
{
    __shared__ alignas(16) __bf16 As[BM * BK];   // [m][k]
    __shared__ alignas(16) __bf16 Bs[BN * BK];   // [n][k]  (transposed W tile)

    const int tid  = threadIdx.x;
    const int wave = tid >> 5;
    const int lane = tid & 31;
    const int wm   = wave & 3;        // 4 waves along M
    const int wn   = wave >> 2;       // 2 waves along N
    const int g    = lane >> 4;       // half-wave group (0/1)
    const int l16  = lane & 15;

    const int rowBase = blockIdx.x * BM;
    const int colBase = blockIdx.y * BN;

    // Per-thread staging coordinates (branchless; row clamped so padded rows
    // load valid memory; their C rows are never stored).
    int aRow[4], aKk[4], aClampedRow[4];
    #pragma unroll
    for (int v = 0; v < 4; v++) {
        int e = (tid + v * 256) * 4;          // [0, 4096)
        aRow[v] = e >> 5;                     // tile row
        aKk[v]  = e & 31;                     // tile k
        int gr = rowBase + aRow[v];
        aClampedRow[v] = (gr < M) ? gr : (M - 1);
    }
    int bK[2], bN[2];
    #pragma unroll
    for (int v = 0; v < 2; v++) {
        int e = (tid + v * 256) * 4;          // [0, 2048)
        bK[v] = e >> 6;                       // tile k
        bN[v] = e & 63;                       // tile n
    }

    v4f aR[4], bR[2];
    auto loadTiles = [&](int k0) {
        #pragma unroll
        for (int v = 0; v < 4; v++)
            aR[v] = *(const v4f*)&A[(size_t)aClampedRow[v] * K + k0 + aKk[v]];
        #pragma unroll
        for (int v = 0; v < 2; v++)
            bR[v] = *(const v4f*)&W[(size_t)(k0 + bK[v]) * N + colBase + bN[v]];
    };
    auto storeTiles = [&]() {
        #pragma unroll
        for (int v = 0; v < 4; v++) {
            v4bf h;
            #pragma unroll
            for (int i = 0; i < 4; i++) h[i] = (__bf16)aR[v][i];
            *(v4bf*)&As[aRow[v] * BK + aKk[v]] = h;     // ds_store_b64
        }
        #pragma unroll
        for (int v = 0; v < 2; v++) {
            #pragma unroll
            for (int i = 0; i < 4; i++)
                Bs[(bN[v] + i) * BK + bK[v]] = (__bf16)bR[v][i];  // transpose
        }
    };

    v8f acc[2][2] = {};

    loadTiles(0);
    for (int k0 = 0; k0 < K; k0 += BK) {
        storeTiles();
        __syncthreads();
        if (k0 + BK < K) loadTiles(k0 + BK);   // overlap with WMMA below

        v16bf afr[2], bfr[2];
        #pragma unroll
        for (int i = 0; i < 2; i++) {
            int m = wm * 32 + i * 16 + l16;       // A: M = lane%16
            v8bf lo = *(const v8bf*)&As[m * BK + g * 8];        // K = g*8 + 0..7
            v8bf hi = *(const v8bf*)&As[m * BK + 16 + g * 8];   // K = 16+g*8 + 0..7
            afr[i] = __builtin_shufflevector(lo, hi, 0,1,2,3,4,5,6,7,8,9,10,11,12,13,14,15);
        }
        #pragma unroll
        for (int j = 0; j < 2; j++) {
            int n = wn * 32 + j * 16 + l16;       // B: N = lane%16
            v8bf lo = *(const v8bf*)&Bs[n * BK + g * 8];        // K = g*8 + 0..7
            v8bf hi = *(const v8bf*)&Bs[n * BK + 16 + g * 8];   // K = 16+g*8 + 0..7
            bfr[j] = __builtin_shufflevector(lo, hi, 0,1,2,3,4,5,6,7,8,9,10,11,12,13,14,15);
        }
        #pragma unroll
        for (int i = 0; i < 2; i++)
            #pragma unroll
            for (int j = 0; j < 2; j++)
                acc[i][j] = __builtin_amdgcn_wmma_f32_16x16x32_bf16(
                    false, afr[i], false, bfr[j], (short)0, acc[i][j], false, false);
        __syncthreads();
    }

    // --- epilogue ---
    #pragma unroll
    for (int i = 0; i < 2; i++) {
        #pragma unroll
        for (int j = 0; j < 2; j++) {
            int col = colBase + wn * 32 + j * 16 + l16;
            #pragma unroll
            for (int r = 0; r < 8; r++) {
                int row = rowBase + wm * 32 + i * 16 + r + g * 8;  // C: M = r + g*8
                if (row < M) {
                    float v = acc[i][j][r];
                    if (bias) v += bias[col];
                    if (flags & 1) {
                        float x = v;
                        float t = tanhf(0.79788456080286536f * (x + 0.044715f * x * x * x));
                        v = 0.5f * x * (1.0f + t);
                    }
                    size_t o = (size_t)row * N + col;
                    if (resid) v = resid[o] + (gate ? gate[col] : 1.0f) * v;
                    C[o] = v;
                }
            }
        }
    }
}

// ---------------------------------------------------------------------------
// RoPE table build. pos = center + scale*grid(G) (local) or grid(G) (scene).
// ---------------------------------------------------------------------------
__global__ void k_rope_build(const float* __restrict__ centers,
                             const float* __restrict__ scales,
                             float* __restrict__ sinb, float* __restrict__ cosb,
                             int G, int B)
{
    int idx = blockIdx.x * blockDim.x + threadIdx.x;
    int total = B * G * G;
    if (idx >= total) return;
    int b = idx / (G * G);
    int n = idx - b * (G * G);
    int iy = n / G, ix = n - iy * G;
    float cy = (iy + 0.5f) / (float)G * 2.0f - 1.0f;
    float cx = (ix + 0.5f) / (float)G * 2.0f - 1.0f;
    float py = cy, px = cx;
    if (centers) {
        py = centers[b * 2 + 0] + scales[b] * cy;
        px = centers[b * 2 + 1] + scales[b] * cx;
    }
    #pragma unroll
    for (int j = 0; j < 16; j++) {
        float per = powf(10000.0f, (float)j / 16.0f);
        float a0 = py / per;
        float a1 = px / per;
        sinb[(size_t)idx * 32 + j]      = sinf(a0);
        cosb[(size_t)idx * 32 + j]      = cosf(a0);
        sinb[(size_t)idx * 32 + 16 + j] = sinf(a1);
        cosb[(size_t)idx * 32 + 16 + j] = cosf(a1);
    }
}

// ---------------------------------------------------------------------------
// Apply RoPE in place to X[M rows, ld stride], per head 64-dim halves of 32.
// ---------------------------------------------------------------------------
__global__ void k_rope_apply(float* __restrict__ X, int ld, int M, int Npos,
                             int batchStride,
                             const float* __restrict__ sinb,
                             const float* __restrict__ cosb)
{
    int t = blockIdx.x * blockDim.x + threadIdx.x;
    int total = M * NH * 32;
    if (t >= total) return;
    int row = t / (NH * 32);
    int r   = t - row * (NH * 32);
    int h = r >> 5;
    int d = r & 31;
    int b = row / Npos;
    int n = row - b * Npos;
    size_t ridx = (size_t)b * batchStride + (size_t)n * 32 + d;
    float s = sinb[ridx], c = cosb[ridx];
    float* p = X + (size_t)row * ld + h * HD;
    float x1 = p[d];
    float x2 = p[d + 32];
    p[d]      = x1 * c - x2 * s;
    p[d + 32] = x2 * c + x1 * s;
}

// ---------------------------------------------------------------------------
// Attention per (batch, head): O = softmax(Q K^T / sqrt(64)) V
// One workgroup (8 waves) per (b,h). K/V head slice staged in LDS with
// async global->LDS b128 copies (ASYNCcnt path) when available.
// ---------------------------------------------------------------------------
__global__ __launch_bounds__(256) void k_attn(
    const float* __restrict__ Q, int ldq,
    const float* __restrict__ K, int ldk,
    const float* __restrict__ V, int ldv,
    float* __restrict__ O, int ldo,
    int Nq, int Nk)
{
    __shared__ alignas(16) float sK[NS * HD];
    __shared__ alignas(16) float sV[NS * HD];
    __shared__ float sS[8 * NS];
    __shared__ float sQ[8 * HD];

    const int tid  = threadIdx.x;
    const int wave = tid >> 5;
    const int lane = tid & 31;
    const int b = blockIdx.x / NH;
    const int h = blockIdx.x % NH;
    const float scale = 0.125f; // 1/sqrt(64)

    for (int e = tid * 4; e < Nk * HD; e += 256 * 4) {
        int j = e >> 6, d = e & 63;
        copy16_g2l(&sK[e], &K[((size_t)b * Nk + j) * ldk + h * HD + d]);
        copy16_g2l(&sV[e], &V[((size_t)b * Nk + j) * ldv + h * HD + d]);
    }
    async_wait_all();
    __syncthreads();

    for (int qi = wave; qi < Nq; qi += 8) {
        // stage q row (wave-local; LDS ops in-order within a wave)
        const float* qrow = Q + ((size_t)b * Nq + qi) * ldq + h * HD;
        for (int d = lane; d < HD; d += 32) sQ[wave * HD + d] = qrow[d];

        // scores
        float mloc = -3.0e38f;
        for (int j = lane; j < Nk; j += 32) {
            float s = 0.0f;
            #pragma unroll
            for (int d = 0; d < HD; d++) s += sQ[wave * HD + d] * sK[j * HD + d];
            s *= scale;
            sS[wave * NS + j] = s;
            mloc = fmaxf(mloc, s);
        }
        #pragma unroll
        for (int off = 16; off > 0; off >>= 1)
            mloc = fmaxf(mloc, __shfl_xor(mloc, off, 32));

        float sum = 0.0f;
        for (int j = lane; j < Nk; j += 32) {
            float e = expf(sS[wave * NS + j] - mloc);
            sS[wave * NS + j] = e;
            sum += e;
        }
        #pragma unroll
        for (int off = 16; off > 0; off >>= 1)
            sum += __shfl_xor(sum, off, 32);
        float inv = 1.0f / sum;

        // output: lanes split the 64 dims (2 each)
        for (int d = lane; d < HD; d += 32) {
            float acc = 0.0f;
            for (int j = 0; j < Nk; j++) acc += sS[wave * NS + j] * sV[j * HD + d];
            O[((size_t)b * Nq + qi) * ldo + h * HD + d] = acc * inv;
        }
    }
}

// ---------------------------------------------------------------------------
// LayerNorm: one wave per row of 768.
// ---------------------------------------------------------------------------
__global__ __launch_bounds__(256) void k_layernorm(
    const float* __restrict__ X, float* __restrict__ Y,
    const float* __restrict__ gm, const float* __restrict__ bt, int M)
{
    int row = blockIdx.x * 8 + (threadIdx.x >> 5);
    int lane = threadIdx.x & 31;
    if (row >= M) return;
    const float* x = X + (size_t)row * D;

    float s = 0.0f;
    for (int i = lane; i < D; i += 32) s += x[i];
    #pragma unroll
    for (int off = 16; off > 0; off >>= 1) s += __shfl_xor(s, off, 32);
    float mean = s * (1.0f / D);

    float v = 0.0f;
    for (int i = lane; i < D; i += 32) { float d0 = x[i] - mean; v += d0 * d0; }
    #pragma unroll
    for (int off = 16; off > 0; off >>= 1) v += __shfl_xor(v, off, 32);
    float rstd = rsqrtf(v * (1.0f / D) + 1e-6f);

    float* y = Y + (size_t)row * D;
    for (int i = lane; i < D; i += 32)
        y[i] = (x[i] - mean) * rstd * gm[i] + bt[i];
}

// ---------------------------------------------------------------------------
// Host driver
// ---------------------------------------------------------------------------
extern "C" void kernel_launch(void* const* d_in, const int* in_sizes, int n_in,
                              void* d_out, int out_size, void* d_ws, size_t ws_size,
                              hipStream_t stream)
{
    const float* in_local  = (const float*)d_in[0];
    const float* in_hidden = (const float*)d_in[1];
    const float* centers   = (const float*)d_in[2];
    const float* scales    = (const float*)d_in[3];
    const float* read_Wq   = (const float*)d_in[4];
    const float* read_Wk   = (const float*)d_in[5];
    const float* read_Wv   = (const float*)d_in[6];
    const float* read_Wo   = (const float*)d_in[7];
    const float* read_gate = (const float*)d_in[8];
    const float* write_Wq  = (const float*)d_in[9];
    const float* write_Wk  = (const float*)d_in[10];
    const float* write_Wv  = (const float*)d_in[11];
    const float* write_Wo  = (const float*)d_in[12];
    const float* write_gate= (const float*)d_in[13];
    const float* ln1_g     = (const float*)d_in[14];
    const float* ln1_b     = (const float*)d_in[15];
    const float* attn_Wqkv = (const float*)d_in[16];
    const float* attn_Wo   = (const float*)d_in[17];
    const float* ln2_g     = (const float*)d_in[18];
    const float* ln2_b     = (const float*)d_in[19];
    const float* mlp_W1    = (const float*)d_in[20];
    const float* mlp_b1    = (const float*)d_in[21];
    const float* mlp_W2    = (const float*)d_in[22];
    const float* mlp_b2    = (const float*)d_in[23];

    float* ws = (float*)d_ws;
    size_t off = 0;
    float* local  = ws + off; off += (size_t)ML * D;
    float* hidden = ws + off; off += (size_t)MH * D;
    float* lsin   = ws + off; off += (size_t)ML * 32;
    float* lcos   = ws + off; off += (size_t)ML * 32;
    float* ssin   = ws + off; off += (size_t)NS * 32;
    float* scos   = ws + off; off += (size_t)NS * 32;
    float* T0     = ws + off; off += (size_t)MH * D;
    float* T1     = ws + off; off += (size_t)MH * D;
    float* T2     = ws + off; off += (size_t)MH * D;
    float* T3     = ws + off; off += (size_t)MH * D;
    float* T4     = ws + off; off += (size_t)ML * 4 * D;

    (void)hipMemcpyAsync(local,  in_local,  (size_t)ML * D * sizeof(float), hipMemcpyDeviceToDevice, stream);
    (void)hipMemcpyAsync(hidden, in_hidden, (size_t)MH * D * sizeof(float), hipMemcpyDeviceToDevice, stream);

    k_rope_build<<<(B_ * NL + 255) / 256, 256, 0, stream>>>(centers, scales, lsin, lcos, 7, B_);
    k_rope_build<<<(NS + 255) / 256, 256, 0, stream>>>(nullptr, nullptr, ssin, scos, 16, 1);

    auto gemm = [&](const float* A, const float* W, float* C, int M, int N, int K,
                    const float* bias, const float* gate, const float* resid, int flags) {
        dim3 grid((M + BM - 1) / BM, N / BN);
        k_gemm_bf16<<<grid, 256, 0, stream>>>(A, W, C, bias, gate, resid, flags, M, N, K);
    };
    auto rope = [&](float* X, int ld, int M, int Npos, int bstride,
                    const float* sb, const float* cb) {
        int total = M * NH * 32;
        k_rope_apply<<<(total + 255) / 256, 256, 0, stream>>>(X, ld, M, Npos, bstride, sb, cb);
    };
    auto attn = [&](const float* Q, int ldq, const float* K, int ldk,
                    const float* V, int ldv, float* O, int Nq, int Nk) {
        k_attn<<<B_ * NH, 256, 0, stream>>>(Q, ldq, K, ldk, V, ldv, O, D, Nq, Nk);
    };
    auto lnorm = [&](const float* X, float* Y, const float* g, const float* b, int M) {
        k_layernorm<<<(M + 7) / 8, 256, 0, stream>>>(X, Y, g, b, M);
    };

    const size_t WD = (size_t)D * D;
    for (int i = 0; i < LNUM; i++) {
        // ---- read cross-attn: local += read_gate * CA(local <- hidden) ----
        gemm(local,  read_Wq + i * WD, T0, ML, D, D, nullptr, nullptr, nullptr, 0);
        rope(T0, D, ML, NL, NL * 32, lsin, lcos);
        gemm(hidden, read_Wk + i * WD, T1, MH, D, D, nullptr, nullptr, nullptr, 0);
        rope(T1, D, MH, NS, 0, ssin, scos);
        gemm(hidden, read_Wv + i * WD, T2, MH, D, D, nullptr, nullptr, nullptr, 0);
        attn(T0, D, T1, D, T2, D, T3, NL, NS);
        gemm(T3, read_Wo + i * WD, local, ML, D, D, nullptr, read_gate + i * D, local, 0);

        // ---- self-attn on local ----
        lnorm(local, T3, ln1_g + i * D, ln1_b + i * D, ML);
        gemm(T3, attn_Wqkv + (size_t)i * D * 3 * D, T0, ML, 3 * D, D, nullptr, nullptr, nullptr, 0);
        rope(T0,     3 * D, ML, NL, NL * 32, lsin, lcos);       // q slice
        rope(T0 + D, 3 * D, ML, NL, NL * 32, lsin, lcos);       // k slice
        attn(T0, 3 * D, T0 + D, 3 * D, T0 + 2 * D, 3 * D, T1, NL, NL);
        gemm(T1, attn_Wo + i * WD, local, ML, D, D, nullptr, nullptr, local, 0);

        // ---- MLP ----
        lnorm(local, T3, ln2_g + i * D, ln2_b + i * D, ML);
        gemm(T3, mlp_W1 + (size_t)i * D * 4 * D, T4, ML, 4 * D, D,
             mlp_b1 + (size_t)i * 4 * D, nullptr, nullptr, 1 /*gelu*/);
        gemm(T4, mlp_W2 + (size_t)i * 4 * D * D, local, ML, D, 4 * D,
             mlp_b2 + i * D, nullptr, local, 0);

        // ---- write cross-attn: hidden += write_gate * CA(hidden <- local) ----
        gemm(hidden, write_Wq + i * WD, T0, MH, D, D, nullptr, nullptr, nullptr, 0);
        rope(T0, D, MH, NS, 0, ssin, scos);
        gemm(local,  write_Wk + i * WD, T1, ML, D, D, nullptr, nullptr, nullptr, 0);
        rope(T1, D, ML, NL, NL * 32, lsin, lcos);
        gemm(local,  write_Wv + i * WD, T2, ML, D, D, nullptr, nullptr, nullptr, 0);
        attn(T0, D, T1, D, T2, D, T3, NS, NL);
        gemm(T3, write_Wo + i * WD, hidden, MH, D, D, nullptr, write_gate + i * D, hidden, 0);
    }

    (void)hipMemcpyAsync(d_out, hidden, (size_t)MH * D * sizeof(float), hipMemcpyDeviceToDevice, stream);
}
